// MSFastSampler_24816321036789
// MI455X (gfx1250) — matrix operand order repaired and checked by reading, not compiled, not run on hardware
//
#include <hip/hip_runtime.h>
#include <hip/hip_bf16.h>

// Problem constants (match reference)
constexpr int Bn   = 256;
constexpr int Dn   = 4096;
constexpr int MAXR = 15;   // 2*R - 1

// Register tiling for the WMMA GEMM: each wave computes (16*MT) x (16*NT)
constexpr int MT = 2;      // m (batch) tiles per wave  -> 32 rows
constexpr int NT = 4;      // n (k-index) tiles per wave -> 64 cols

typedef __attribute__((ext_vector_type(16))) __bf16 v16bf;
typedef __attribute__((ext_vector_type(8)))  __bf16 v8bf;
typedef __attribute__((ext_vector_type(8)))  float  v8f;

// ---------------------------------------------------------------------------
// Kernel 1: Wsym[k,j] = 0.5*(W[k,j] + W[j,k]) in bf16 (folds the 0.5 factor).
// LDS-tiled so BOTH the direct and transposed reads are coalesced 64B lines.
// 64x64 tile per 256-thread block; padded LDS (65) avoids bank conflicts.
// ---------------------------------------------------------------------------
__global__ __launch_bounds__(256) void k_symmetrize(const float* __restrict__ W,
                                                    __bf16* __restrict__ Ws) {
    __shared__ float T1[64][65];
    __shared__ float T2[64][65];
    const int bj = blockIdx.x;   // column tile
    const int bi = blockIdx.y;   // row tile
    const int t  = threadIdx.x;

#pragma unroll
    for (int q = 0; q < 16; ++q) {
        int lin = q * 256 + t;
        int r = lin >> 6, c = lin & 63;
        T1[r][c] = W[(size_t)(bi * 64 + r) * Dn + bj * 64 + c];  // W[bi,bj] tile
        T2[r][c] = W[(size_t)(bj * 64 + r) * Dn + bi * 64 + c];  // W[bj,bi] tile
    }
    __syncthreads();
#pragma unroll
    for (int q = 0; q < 16; ++q) {
        int lin = q * 256 + t;
        int r = lin >> 6, c = lin & 63;
        float v = 0.5f * (T1[r][c] + T2[c][r]);                  // W + W^T
        Ws[(size_t)(bi * 64 + r) * Dn + bj * 64 + c] = (__bf16)v;
    }
}

// ---------------------------------------------------------------------------
// Kernel 2: x (binary fp32) -> bf16 (exact)
// ---------------------------------------------------------------------------
__global__ void k_convert_x(const float* __restrict__ x, __bf16* __restrict__ xb) {
    size_t i = (size_t)blockIdx.x * blockDim.x + threadIdx.x;
    if (i >= (size_t)Bn * Dn) return;
    xb[i] = (__bf16)x[i];
}

// ---------------------------------------------------------------------------
// Kernel 3: register-tiled WMMA GEMM.
//   grad[b,k] = sum_j Wsym[k,j] * x[b,j] + bias[k]
//   scx[b,k]  = 0.5 * (1 - 2*x[b,k]) * grad[b,k]
// One wave per 32x64 output block: MT x NT = 8 accumulators; per K-step of 32
// it loads MT A-fragments + NT B-fragments and issues 8 WMMAs (~2.7x better
// compute/load ratio than 1x1 tiling, ~2.7x less L2 traffic).
// Fragment layouts per CDNA5 ISA 7.12.2 (bf16 A 16x32, B 32x16, C/D f32 16x16):
//   A lane L (m = L&15, half = L>>4): VGPR0-3 = K (8*half)+0..7, VGPR4-7 = K 16+8*half..+7
//   B lane L (n = L&15, half = L>>4): VGPR0-3 = K (8*half)+0..7, VGPR4-7 = K 16+8*half..+7, column n
//   C lane L (n = L&15): VGPR v = row v + 8*(L>>4)
// All fragment pieces are contiguous 16-byte chunks -> global_load_b128.
// ---------------------------------------------------------------------------
__global__ __launch_bounds__(32) void k_grad_gemm(const __bf16* __restrict__ Xb,
                                                  const __bf16* __restrict__ Ws,
                                                  const float*  __restrict__ xf,
                                                  const float*  __restrict__ bias,
                                                  float* __restrict__ grad,
                                                  float* __restrict__ scx) {
    const int lane = threadIdx.x & 31;
    const int half = lane >> 4;
    const int l15  = lane & 15;
    const int n0   = blockIdx.x * (16 * NT);   // output column block (k)
    const int m0   = blockIdx.y * (16 * MT);   // batch row block (b)

    const __bf16* arow[MT];
    const __bf16* brow[NT];
#pragma unroll
    for (int mt = 0; mt < MT; ++mt)
        arow[mt] = Xb + (size_t)(m0 + mt * 16 + l15) * Dn;
#pragma unroll
    for (int nt = 0; nt < NT; ++nt)
        brow[nt] = Ws + (size_t)(n0 + nt * 16 + l15) * Dn;

    v8f acc[MT][NT];
#pragma unroll
    for (int mt = 0; mt < MT; ++mt)
#pragma unroll
        for (int nt = 0; nt < NT; ++nt)
            acc[mt][nt] = (v8f){};

    for (int j0 = 0; j0 < Dn; j0 += 32) {
        v16bf av[MT], bv[NT];
#pragma unroll
        for (int mt = 0; mt < MT; ++mt) {
            v8bf a0 = *(const v8bf*)(arow[mt] + j0 + 8 * half);
            v8bf a1 = *(const v8bf*)(arow[mt] + j0 + 16 + 8 * half);
#pragma unroll
            for (int t = 0; t < 8; ++t) { av[mt][t] = a0[t]; av[mt][8 + t] = a1[t]; }
        }
#pragma unroll
        for (int nt = 0; nt < NT; ++nt) {
            v8bf b0 = *(const v8bf*)(brow[nt] + j0 + 8 * half);
            v8bf b1 = *(const v8bf*)(brow[nt] + j0 + 16 + 8 * half);
#pragma unroll
            for (int t = 0; t < 8; ++t) { bv[nt][t] = b0[t]; bv[nt][8 + t] = b1[t]; }
        }
#pragma unroll
        for (int mt = 0; mt < MT; ++mt)
#pragma unroll
            for (int nt = 0; nt < NT; ++nt)
                acc[mt][nt] = __builtin_amdgcn_wmma_f32_16x16x32_bf16(
                    false, av[mt], false, bv[nt], (short)0, acc[mt][nt],
                    false, false);
    }

#pragma unroll
    for (int nt = 0; nt < NT; ++nt) {
        const int   col = n0 + nt * 16 + l15;
        const float bk  = bias[col];
#pragma unroll
        for (int mt = 0; mt < MT; ++mt) {
#pragma unroll
            for (int v = 0; v < 8; ++v) {
                int    row = m0 + mt * 16 + v + 8 * half;
                size_t o   = (size_t)row * Dn + col;
                float  gv  = acc[mt][nt][v] + bk;
                grad[o] = gv;
                float delta = 1.0f - 2.0f * xf[o];       // -(2x-1)
                scx[o]  = 0.5f * delta * gv;
            }
        }
    }
}

// ---------------------------------------------------------------------------
// Kernel 4: per-row sampler. One 256-thread block per batch row.
//   - Gumbel keys, iterative top-15 argmax (16KB LDS key array)
//   - incremental score_y - score_x = d.grad + 0.5 d^T W d
//   - logsumexp(scy) from logsumexp(scx) via sign-flip adjustment at flips
//   - MH accept, write output row
// ---------------------------------------------------------------------------
__global__ void k_sample(const float* __restrict__ x,
                         const float* __restrict__ W,
                         const int*   __restrict__ radius_raw,
                         const float* __restrict__ gum,
                         const float* __restrict__ u,
                         const float* __restrict__ grad,
                         const float* __restrict__ scx,
                         float* __restrict__ out) {
    __shared__ float s_keys[Dn];      // 16 KB
    __shared__ float s_rv[256];
    __shared__ int   s_ri[256];
    __shared__ int   s_idx[MAXR];
    __shared__ float s_m, s_Zx;
    __shared__ int   s_acc, s_r;

    const int row = blockIdx.x;
    const int t   = threadIdx.x;
    const float* scxr = scx  + (size_t)row * Dn;
    const float* xr   = x    + (size_t)row * Dn;
    const float* gr   = grad + (size_t)row * Dn;
    const float* gur  = gum  + (size_t)row * Dn;
    const float NEG_INF = -__builtin_inff();

    // Gumbel keys + row max of scx
    float lmax = NEG_INF;
    for (int j = t; j < Dn; j += 256) {
        float s  = scxr[j];
        float uu = fminf(fmaxf(gur[j], 1e-10f), 1.0f - 1e-10f);
        float g  = -__logf(-__logf(uu));
        s_keys[j] = s + g;
        lmax = fmaxf(lmax, s);
    }
    s_rv[t] = lmax;
    __syncthreads();
    for (int off = 128; off > 0; off >>= 1) {
        if (t < off) s_rv[t] = fmaxf(s_rv[t], s_rv[t + off]);
        __syncthreads();
    }
    if (t == 0) s_m = s_rv[0];
    __syncthreads();
    const float m = s_m;

    // sum exp(scx - m)
    float lsum = 0.0f;
    for (int j = t; j < Dn; j += 256) lsum += __expf(scxr[j] - m);
    s_rv[t] = lsum;
    __syncthreads();
    for (int off = 128; off > 0; off >>= 1) {
        if (t < off) s_rv[t] += s_rv[t + off];
        __syncthreads();
    }
    if (t == 0) s_Zx = s_rv[0];
    __syncthreads();

    // iterative top-15 argmax over keys
    for (int it = 0; it < MAXR; ++it) {
        float bv = NEG_INF;
        int   bi = 0;
        for (int j = t; j < Dn; j += 256) {
            float v = s_keys[j];
            if (v > bv) { bv = v; bi = j; }
        }
        s_rv[t] = bv; s_ri[t] = bi;
        __syncthreads();
        for (int off = 128; off > 0; off >>= 1) {
            if (t < off && s_rv[t + off] > s_rv[t]) {
                s_rv[t] = s_rv[t + off];
                s_ri[t] = s_ri[t + off];
            }
            __syncthreads();
        }
        if (t == 0) {
            s_idx[it] = s_ri[0];
            s_keys[s_ri[0]] = NEG_INF;
        }
        __syncthreads();
    }

    // scalar accept logic (thread 0)
    if (t == 0) {
        int r = radius_raw[row] + 1;
        if (r < 1) r = 1;
        if (r > MAXR) r = MAXR;
        s_r = r;

        float dsign[MAXR];
        float sum_dg = 0.0f;
        float m2 = m;
        for (int i = 0; i < r; ++i) {
            int id = s_idx[i];
            float d = 1.0f - 2.0f * xr[id];     // y - x at flip
            dsign[i] = d;
            sum_dg += d * gr[id];
            m2 = fmaxf(m2, -scxr[id]);
        }
        // 0.5 * d^T W d (gather from original fp32 W; d^T W d == d^T Wsym d)
        float quad = 0.0f;
        for (int i = 0; i < r; ++i) {
            const float* Wr = W + (size_t)s_idx[i] * Dn;
            for (int j2 = 0; j2 < r; ++j2)
                quad += dsign[i] * dsign[j2] * Wr[s_idx[j2]];
        }
        quad *= 0.5f;

        float Zx  = s_Zx;
        float Zx2 = Zx * __expf(m - m2);
        float adj = 0.0f;
        for (int i = 0; i < r; ++i) {
            float sc = scxr[s_idx[i]];
            adj += __expf(-sc - m2) - __expf(sc - m2);   // flipped entry sign change
        }
        float Zy2 = Zx2 + adj;
        if (!(Zy2 > 1e-38f)) Zy2 = 1e-38f;

        float log_zxzy = (m + __logf(Zx)) - (m2 + __logf(Zy2));
        float log_pypx  = sum_dg + quad;      // score_y - score_x
        float log_tilde = -sum_dg;            // -sum(grad*(y-x))
        float log_acc   = fminf(log_pypx + log_tilde + log_zxzy, 0.0f);
        s_acc = (__expf(log_acc) > u[row]) ? 1 : 0;
    }
    __syncthreads();

    // write output row
    float* outr = out + (size_t)row * Dn;
    for (int j = t; j < Dn; j += 256) outr[j] = xr[j];
    __syncthreads();
    if (s_acc && t < s_r) {
        int id = s_idx[t];
        outr[id] = 1.0f - xr[id];
    }
}

// ---------------------------------------------------------------------------
extern "C" void kernel_launch(void* const* d_in, const int* in_sizes, int n_in,
                              void* d_out, int out_size, void* d_ws, size_t ws_size,
                              hipStream_t stream) {
    (void)in_sizes; (void)n_in; (void)out_size; (void)ws_size;

    const float* x          = (const float*)d_in[0];   // [B,D]
    const float* W          = (const float*)d_in[1];   // [D,D]
    const float* bias       = (const float*)d_in[2];   // [D]
    const int*   radius_raw = (const int*)  d_in[3];   // [B,1]
    const float* gumbel_u   = (const float*)d_in[4];   // [B,D]
    const float* u          = (const float*)d_in[5];   // [B]
    float*       out        = (float*)d_out;           // [B,D]

    // workspace layout
    char*   ws     = (char*)d_ws;
    size_t  offWs  = 0;
    size_t  offXb  = offWs  + (size_t)Dn * Dn * 2;     // Wsym bf16: 32 MB
    size_t  offG   = offXb  + (size_t)Bn * Dn * 2;     // x bf16:    2 MB
    size_t  offScx = offG   + (size_t)Bn * Dn * 4;     // grad f32:  4 MB
    __bf16* Wsym = (__bf16*)(ws + offWs);
    __bf16* Xb   = (__bf16*)(ws + offXb);
    float*  grad = (float*) (ws + offG);
    float*  scx  = (float*) (ws + offScx);             // scx f32:   4 MB

    // 1. symmetrize W -> bf16 (LDS-tiled transpose, fully coalesced)
    {
        dim3 grid(Dn / 64, Dn / 64);
        k_symmetrize<<<grid, 256, 0, stream>>>(W, Wsym);
    }
    // 2. x -> bf16
    {
        size_t total = (size_t)Bn * Dn;
        int blocks = (int)((total + 255) / 256);
        k_convert_x<<<blocks, 256, 0, stream>>>(x, Xb);
    }
    // 3. register-tiled WMMA GEMM: grad + score_change_x
    {
        dim3 grid(Dn / (16 * NT), Bn / (16 * MT));   // 64 x 8 = 512 waves
        k_grad_gemm<<<grid, 32, 0, stream>>>(Xb, Wsym, x, bias, grad, scx);
    }
    // 4. per-row sampler
    {
        k_sample<<<Bn, 256, 0, stream>>>(x, W, radius_raw, gumbel_u, u, grad, scx, out);
    }
}
